// SelfAttention_26886495273176
// MI455X (gfx1250) — matrix operand order repaired and checked by reading, not compiled, not run on hardware
//
#include <hip/hip_runtime.h>

typedef __attribute__((ext_vector_type(16))) _Float16 v16h;
typedef __attribute__((ext_vector_type(8)))  float    v8f;
typedef __attribute__((ext_vector_type(4)))  unsigned int v4u;
typedef __attribute__((ext_vector_type(8)))  int      v8i_;
typedef __attribute__((ext_vector_type(4)))  int      v4i_;

constexpr int S_LEN = 2048;
constexpr int NHEAD = 16;
constexpr int DHEAD = 64;
constexpr int QKV_S_STRIDE = 3 * NHEAD * DHEAD;   // 3072 floats per (b, s)
constexpr int WAVES_PER_BLOCK = 4;
constexpr int QTILE = 32;                         // query rows per wave (2 x 16-row frags)

// Flash-attention forward, one wave32 per 32-query tile of one (b, h).
// QK^T / PV via v_wmma_f32_16x16x32_f16 (16 per 32-key step); the strided
// 32x64 f32 V tile is staged into LDS by the Tensor Data Mover
// (tensor_load_to_lds, TENSORcnt) overlapping the score/softmax phase.
__global__ __launch_bounds__(32 * WAVES_PER_BLOCK)
void fa_fwd_wmma(const float* __restrict__ qkv, float* __restrict__ out) {
  const int lane = threadIdx.x & 31;
  const int wave = threadIdx.x >> 5;
  const int tile = blockIdx.x * WAVES_PER_BLOCK + wave;   // 0 .. B*H*(S/32)-1

  const int qb = tile % (S_LEN / QTILE);
  const int bh = tile / (S_LEN / QTILE);
  const int h  = bh % NHEAD;
  const int b  = bh / NHEAD;
  const int q0 = qb * QTILE;

  const int hw    = lane >> 4;        // half-wave: 0 or 1
  const int l16   = lane & 15;
  const int kbase = hw * 8;           // A-fragment packed-K base

  // per-wave LDS: P (32x32 f16) transpose scratch + V (32x64 f32) TDM stage
  __shared__ _Float16 lds_p[WAVES_PER_BLOCK][QTILE * 32 + 8];
  __shared__ float    lds_v[WAVES_PER_BLOCK][32 * DHEAD];
  _Float16* Pst = &lds_p[wave][0];
  const float* Vl = &lds_v[wave][0];

  const float scale = 0.125f;         // 1/sqrt(64)

  // ---- Q fragments: A-layout 16x32 f16; [query-subtile][d-chunk]
  v16h aQ[2][2];
#pragma unroll
  for (int qs = 0; qs < 2; ++qs) {
    const float* qrow =
        qkv + ((size_t)b * S_LEN + (q0 + 16 * qs + l16)) * QKV_S_STRIDE + 0 * NHEAD * DHEAD + h * DHEAD;
#pragma unroll
    for (int c = 0; c < 2; ++c)
#pragma unroll
      for (int i = 0; i < 8; ++i) {
        aQ[qs][c][i]     = (_Float16)(qrow[32 * c + kbase + i] * scale);
        aQ[qs][c][i + 8] = (_Float16)(qrow[32 * c + kbase + 16 + i] * scale);
      }
  }

  // ---- wave-uniform TDM descriptor pieces (group1 invariant across steps)
  const unsigned int lds_v_off =
      (unsigned int)__builtin_amdgcn_readfirstlane((int)(unsigned int)(size_t)(void*)&lds_v[wave][0]);
  v8i_ g1;
  g1[0] = 0x00020000;                 // workgroup_mask=0, data_size=2 (4 bytes)
  g1[1] = (int)(64u << 16);           // tensor_dim0[15:0]=64 in bits[31:16]
  g1[2] = (int)(32u << 16);           // tensor_dim0 hi=0; tensor_dim1[15:0]=32
  g1[3] = (int)(64u << 16);           // tensor_dim1 hi=0; tile_dim0=64
  g1[4] = 32;                         // tile_dim1=32, tile_dim2=0
  g1[5] = QKV_S_STRIDE;               // tensor_dim0_stride = 3072 elements
  g1[6] = 0;
  g1[7] = 0;
  const v4i_ gz = {0, 0, 0, 0};
#if __clang_major__ >= 23
  const v8i_ gz8 = {0, 0, 0, 0, 0, 0, 0, 0};
#endif

  // ---- accumulators: O is 32x64 f32 as [qsub][4 d-tiles]; per-row softmax state
  v8f O[2][4];
  float m_i[2][8], l_i[2][8];
#pragma unroll
  for (int qs = 0; qs < 2; ++qs) {
#pragma unroll
    for (int t = 0; t < 4; ++t)
#pragma unroll
      for (int r = 0; r < 8; ++r) O[qs][t][r] = 0.f;
#pragma unroll
    for (int r = 0; r < 8; ++r) { m_i[qs][r] = -1e30f; l_i[qs][r] = 0.f; }
  }

  for (int k0 = 0; k0 < q0 + QTILE; k0 += 32) {
    // ---- kick off async TDM load of the V tile (32 keys x 64 d, f32) to LDS
    {
      const float* vtile =
          qkv + ((size_t)b * S_LEN + k0) * QKV_S_STRIDE + 2 * NHEAD * DHEAD + h * DHEAD;
      const unsigned long long ga = (unsigned long long)(size_t)vtile;
      v4u g0;
      g0[0] = 1u;                                                    // count=1
      g0[1] = lds_v_off;                                             // lds_addr
      g0[2] = (unsigned int)__builtin_amdgcn_readfirstlane((int)(unsigned int)ga);
      g0[3] = (unsigned int)__builtin_amdgcn_readfirstlane(
                  (int)(((unsigned int)(ga >> 32) & 0x01FFFFFFu) | (2u << 30)));  // addr hi + type=2
#if __clang_major__ >= 23
      __builtin_amdgcn_tensor_load_to_lds(g0, g1, gz, gz, gz8, 0);
#else
      __builtin_amdgcn_tensor_load_to_lds(g0, g1, gz, gz, 0);
#endif
    }

    // ---- K fragments (shared by both query subtiles): [key-subtile][d-chunk]
    v16h bK[2][2];
#pragma unroll
    for (int sub = 0; sub < 2; ++sub) {
      const float* krow =
          qkv + ((size_t)b * S_LEN + (k0 + sub * 16 + l16)) * QKV_S_STRIDE + 1 * NHEAD * DHEAD + h * DHEAD;
#pragma unroll
      for (int i = 0; i < 16; ++i) {   // B-frag: element i -> K(d) = 8*hw + i + 8*(i>>3)
        bK[sub][0][i] = (_Float16)krow[8 * hw + i + 8 * (i >> 3)];
        bK[sub][1][i] = (_Float16)krow[32 + 8 * hw + i + 8 * (i >> 3)];
      }
    }

    const bool need_mask = (k0 + 31 > q0);

    // ---- per query subtile: scores, mask, online softmax, stage P in LDS
#pragma unroll
    for (int qs = 0; qs < 2; ++qs) {
      v8f Sacc[2];
#pragma unroll
      for (int sub = 0; sub < 2; ++sub) {
        v8f c;
#pragma unroll
        for (int r = 0; r < 8; ++r) c[r] = 0.f;
        c = __builtin_amdgcn_wmma_f32_16x16x32_f16(false, aQ[qs][0], false, bK[sub][0], (short)0, c, false, false);
        c = __builtin_amdgcn_wmma_f32_16x16x32_f16(false, aQ[qs][1], false, bK[sub][1], (short)0, c, false, false);
        Sacc[sub] = c;
      }

      if (need_mask) {
#pragma unroll
        for (int sub = 0; sub < 2; ++sub) {
          const int key = k0 + sub * 16 + l16;
#pragma unroll
          for (int r = 0; r < 8; ++r) {
            const int query = q0 + 16 * qs + r + 8 * hw;
            if (key > query) Sacc[sub][r] = -1e30f;
          }
        }
      }

#pragma unroll
      for (int r = 0; r < 8; ++r) {
        float mx = fmaxf(Sacc[0][r], Sacc[1][r]);
#pragma unroll
        for (int off = 8; off >= 1; off >>= 1) mx = fmaxf(mx, __shfl_xor(mx, off, 32));
        const float mnew  = fmaxf(m_i[qs][r], mx);
        const float alpha = __expf(m_i[qs][r] - mnew);
        m_i[qs][r] = mnew;
        const float p0 = __expf(Sacc[0][r] - mnew);
        const float p1 = __expf(Sacc[1][r] - mnew);
        Sacc[0][r] = p0;
        Sacc[1][r] = p1;
        float rs = p0 + p1;
#pragma unroll
        for (int off = 8; off >= 1; off >>= 1) rs += __shfl_xor(rs, off, 32);
        l_i[qs][r] = l_i[qs][r] * alpha + rs;
#pragma unroll
        for (int t = 0; t < 4; ++t) O[qs][t][r] *= alpha;
      }

      // C-layout -> LDS (row = query-in-tile, col = key-in-block)
#pragma unroll
      for (int sub = 0; sub < 2; ++sub)
#pragma unroll
        for (int r = 0; r < 8; ++r)
          Pst[(16 * qs + r + 8 * hw) * 32 + sub * 16 + l16] = (_Float16)Sacc[sub][r];
    }

    asm volatile("s_wait_dscnt 0" ::: "memory");

    // ---- reload P as A-fragments (16x32 f16) for both query subtiles
    v16h aP[2];
#pragma unroll
    for (int qs = 0; qs < 2; ++qs)
#pragma unroll
      for (int i = 0; i < 8; ++i) {
        aP[qs][i]     = Pst[(16 * qs + l16) * 32 + kbase + i];
        aP[qs][i + 8] = Pst[(16 * qs + l16) * 32 + kbase + 16 + i];
      }

    // ---- V tile is in LDS once the TDM transfer retires
    __builtin_amdgcn_s_wait_tensorcnt(0);
    asm volatile("" ::: "memory");

    // ---- O += P * V: each bV fragment feeds both query subtiles
#pragma unroll
    for (int t = 0; t < 4; ++t) {
      v16h bV;
#pragma unroll
      for (int i = 0; i < 16; ++i) {   // element i -> K(key) = 8*hw + i + 8*(i>>3)
        bV[i] = (_Float16)Vl[(8 * hw + i + 8 * (i >> 3)) * DHEAD + 16 * t + l16];
      }
      O[0][t] = __builtin_amdgcn_wmma_f32_16x16x32_f16(false, aP[0], false, bV, (short)0, O[0][t], false, false);
      O[1][t] = __builtin_amdgcn_wmma_f32_16x16x32_f16(false, aP[1], false, bV, (short)0, O[1][t], false, false);
    }
  }

  // ---- epilogue: normalize by row sum, write out (b, q, h, d)
#pragma unroll
  for (int qs = 0; qs < 2; ++qs)
#pragma unroll
    for (int t = 0; t < 4; ++t)
#pragma unroll
      for (int r = 0; r < 8; ++r) {
        const int m = 16 * qs + r + 8 * hw;
        const int d = 16 * t + l16;
        out[(((size_t)b * S_LEN + (q0 + m)) * NHEAD + h) * DHEAD + d] = O[qs][t][r] / l_i[qs][r];
      }
}

extern "C" void kernel_launch(void* const* d_in, const int* in_sizes, int n_in,
                              void* d_out, int out_size, void* d_ws, size_t ws_size,
                              hipStream_t stream) {
  const float* qkv = (const float*)d_in[0];
  float* out = (float*)d_out;
  const int B = 2;
  const int tiles = B * NHEAD * (S_LEN / QTILE);       // 2048 query tiles
  dim3 grid(tiles / WAVES_PER_BLOCK);                  // 512 blocks
  dim3 block(32 * WAVES_PER_BLOCK);                    // 4 wave32 per block
  fa_fwd_wmma<<<grid, block, 0, stream>>>(qkv, out);
}
